// MixedTransformerBlock_75892072120442
// MI455X (gfx1250) — matrix operand order repaired and checked by reading, not compiled, not run on hardware
//
#include <hip/hip_runtime.h>
#include <hip/hip_bf16.h>
#include <math.h>

// ---------------------------------------------------------------------------
// MixedTransformerBlock for MI455X (gfx1250, wave32, WMMA).
// NHWC activations -> contiguous b128 staging clauses; double-buffered B tile
// (one barrier per K-step, loads overlap WMMA); LDS-transposed epilogues.
// ---------------------------------------------------------------------------

#define NPIX 65536   // 256*256
#define EPS  1e-5f

typedef _Float16 h16;
typedef __attribute__((ext_vector_type(16))) _Float16 v16h;
typedef __attribute__((ext_vector_type(8)))  _Float16 v8h;
typedef __attribute__((ext_vector_type(4)))  float    v4f;
typedef __attribute__((ext_vector_type(8)))  float    v8f;

__device__ inline v8f zero8() {
  v8f z;
#pragma unroll
  for (int i = 0; i < 8; ++i) z[i] = 0.0f;
  return z;
}

__device__ inline v16h cat8(v8h a, v8h b) {
  return __builtin_shufflevector(a, b, 0, 1, 2, 3, 4, 5, 6, 7,
                                 8, 9, 10, 11, 12, 13, 14, 15);
}

// A fragment: lane (m) reads row[kb..kb+7] and row[16+kb..16+kb+7], kb in {0,8}
__device__ inline v16h loadA(const h16* row, int kb) {
  return cat8(*(const v8h*)(row + kb), *(const v8h*)(row + 16 + kb));
}
// B fragment: lane (n) reads 16 consecutive k at kb in {0,16}
__device__ inline v16h loadB(const h16* row, int kb) {
  return cat8(*(const v8h*)(row + kb), *(const v8h*)(row + kb + 8));
}

// 32 contiguous channels -> h16[32], zeroed when !ok (value select, no branch)
__device__ inline void chunkLoad(const h16* base, bool ok, h16* tb) {
#pragma unroll
  for (int i = 0; i < 4; ++i) {
    v8h v = *(const v8h*)(base + 8 * i);
#pragma unroll
    for (int j = 0; j < 8; ++j) tb[8 * i + j] = ok ? v[j] : (h16)0.f;
  }
}
__device__ inline void chunkLoad(const float* base, bool ok, h16* tb) {
#pragma unroll
  for (int i = 0; i < 8; ++i) {
    v4f v = *(const v4f*)(base + 4 * i);
#pragma unroll
    for (int j = 0; j < 4; ++j) tb[4 * i + j] = ok ? (h16)v[j] : (h16)0.f;
  }
}

// shift_plus source pixel for a channel group (uniform grp -> scalar branches)
__device__ inline void shift_src(int grp, int p, int pw, int ph, int& srcp, bool& ok) {
  srcp = p; ok = true;
  if (grp == 0)      { srcp = p + 1;   ok = (pw != 255); }
  else if (grp == 1) { srcp = p - 1;   ok = (pw != 0);   }
  else if (grp == 2) { srcp = p + 256; ok = (ph != 255); }
  else if (grp == 3) { srcp = p - 256; ok = (ph != 0);   }
}

// ---------------------------------------------------------------------------
// Layout prep kernels
// ---------------------------------------------------------------------------
__global__ void to_nhwc(const float* __restrict__ x, float* __restrict__ y) {
  int p = blockIdx.x * 256 + threadIdx.x;
#pragma unroll 4
  for (int c = 0; c < 180; ++c)
    y[(size_t)p * 192 + c] = x[(size_t)c * NPIX + p];
}

// qkv padded-row map: padded row i (0..383) -> src row / feature
__device__ inline int qrow_src(int i, int* f_out) {
  int branch = i / 192, w = i % 192;
  int qv = w / 96, w96 = w % 96;
  int head = w96 / 16, f = w96 % 16;
  *f_out = f;
  return branch * 180 + qv * 90 + head * 15 + f;
}

__global__ void cvt_qkv(const float* __restrict__ src, const float* __restrict__ g,
                        const float* __restrict__ v, h16* __restrict__ dst) {
  int i = blockIdx.x * 256 + threadIdx.x;  // over 384*192
  if (i < 384 * 192) {
    int drow = i / 192, k = i % 192;
    int f; int srow = qrow_src(drow, &f);
    h16 o = (h16)0.f;
    if (f < 15 && k < 180) {
      float s = g[srow] * rsqrtf(v[srow] + EPS);
      o = (h16)(src[srow * 180 + k] * s);
    }
    dst[i] = o;
  }
}

__global__ void fuse_bias_qkv(const float* __restrict__ b, const float* __restrict__ g,
                              const float* __restrict__ beta, const float* __restrict__ m,
                              const float* __restrict__ v, float* __restrict__ dst) {
  int i = blockIdx.x * 256 + threadIdx.x;
  if (i < 384) {
    int f; int srow = qrow_src(i, &f);
    float o = 0.f;
    if (f < 15) {
      float inv = g[srow] * rsqrtf(v[srow] + EPS);
      o = b[srow] * inv + beta[srow] - m[srow] * inv;
    }
    dst[i] = o;
  }
}

// proj weight: K remapped from head-padded attn-out layout (192 cols)
__global__ void cvt_proj(const float* __restrict__ src, h16* __restrict__ dst) {
  int i = blockIdx.x * 256 + threadIdx.x;  // over 192*192
  if (i < 192 * 192) {
    int drow = i / 192, k = i % 192;
    int branch = k / 96, w = k % 96, head = w / 16, f = w % 16;
    int scol = branch * 90 + head * 15 + f;
    dst[i] = (drow < 180 && f < 15) ? (h16)src[drow * 180 + scol] : (h16)0.f;
  }
}

__global__ void cvt_w_pad(const float* __restrict__ src, h16* __restrict__ dst,
                          int M, int K, int Mpad, int Kpad) {
  int i = blockIdx.x * 256 + threadIdx.x;
  if (i < Mpad * Kpad) {
    int row = i / Kpad, k = i - row * Kpad;
    dst[i] = (row < M && k < K) ? (h16)src[row * K + k] : (h16)0.f;
  }
}

__global__ void pad_bias(const float* __restrict__ src, float* __restrict__ dst,
                         int M, int Mpad) {
  int i = blockIdx.x * 256 + threadIdx.x;
  if (i < Mpad) dst[i] = (i < M) ? src[i] : 0.f;
}

// ---------------------------------------------------------------------------
// WMMA GEMM over NHWC activations: Y[p,co] = sum_ci W[co,ci] * X[p,ci]
//   EPI 0: Yh(NHWC f16) = acc + bias          EPI 1: Yh(NHWC f16) = gelu(...)
//   EPI 2: Yf(NHWC f32) = res + acc + bias    EPI 3: Yf(NCHW f32) = res + ...
//   SHIFT 1: shift_plus gather; a K-step spans <=2 groups (grpSize >= 32).
// Block: 256 thr = 8 waves as 2(M)x4(N); tile 32 x 256 pixels.
// B tile double-buffered in LDS: one barrier per K-step, loads overlap WMMA.
// ---------------------------------------------------------------------------
template <int EPI, int SHIFT, typename TB>
__global__ __launch_bounds__(256)
void gemm_wmma(const h16* __restrict__ Wf,  // Mpad x Kpad, zero padded
               const TB* __restrict__ X,    // NHWC, row stride CSin
               const float* __restrict__ bias,  // Mpad, zero padded
               const float* __restrict__ res,   // NHWC f32, stride CSout(/192)
               float* __restrict__ Yf, h16* __restrict__ Yh,
               int M, int Kpad, int CSin, int CSout, int grpSize) {
  __shared__ alignas(16) unsigned char smem[40960];
  h16 (*Bbuf0)[40] = (h16(*)[40])smem;               // 20.5 KB
  h16 (*Bbuf1)[40] = (h16(*)[40])(smem + 20480);     // 20.5 KB
  h16 (*sTh)[40]   = (h16(*)[40])smem;               // f16 out transpose
  float (*sTf)[36] = (float(*)[36])smem;             // f32 out transpose (36 KB)

  const int tid  = threadIdx.x;
  const int wave = tid >> 5;
  const int lane = tid & 31;
  const int nloc = lane & 15;
  const int hi   = (lane & 16) ? 8 : 0;
  const int kbA  = (lane & 16) ? 8 : 0;
  const int kbB  = (lane & 16) ? 16 : 0;
  const int wm = wave >> 2;
  const int wn = wave & 3;

  const int mBase   = blockIdx.y << 5;
  const int pixBase = blockIdx.x << 8;
  const int mW = mBase + (wm << 4);

  const h16* arow = Wf + (size_t)(mW + nloc) * Kpad + kbA;

  v8f acc[4];
#pragma unroll
  for (int s = 0; s < 4; ++s) acc[s] = zero8();

  const int Ksteps = Kpad >> 5;
  const int p = pixBase + tid;
  const int ph = p >> 8, pw = p & 255;
  const TB* Xrow = X + (size_t)p * CSin;

  // stage one 32-channel K-slab into dst[tid][0..31]
  auto stage = [&](int k0, h16 (*dst)[40]) {
    h16 tb[32];
    if (SHIFT) {
      int grpA = k0 / grpSize;        if (grpA > 4) grpA = 4;
      int grpB = (k0 + 31) / grpSize; if (grpB > 4) grpB = 4;
      int sA; bool okA;
      shift_src(grpA, p, pw, ph, sA, okA);
      chunkLoad(X + (size_t)(okA ? sA : p) * CSin + k0, okA, tb);
      if (grpB != grpA) {             // uniform branch
        int sB; bool okB;
        shift_src(grpB, p, pw, ph, sB, okB);
        h16 tb2[32];
        chunkLoad(X + (size_t)(okB ? sB : p) * CSin + k0, okB, tb2);
        const int bnd = grpB * grpSize - k0;  // 1..31, uniform
#pragma unroll
        for (int r = 0; r < 32; ++r) tb[r] = (r < bnd) ? tb[r] : tb2[r];
      }
    } else {
      chunkLoad(Xrow + k0, true, tb);
    }
#pragma unroll
    for (int i = 0; i < 4; ++i)
      *(v8h*)&dst[tid][8 * i] = *(const v8h*)&tb[8 * i];
  };

  stage(0, Bbuf0);
  __syncthreads();

  for (int ks = 0; ks < Ksteps; ++ks) {
    const int k0 = ks << 5;
    h16 (*cur)[40] = (ks & 1) ? Bbuf1 : Bbuf0;
    h16 (*nxt)[40] = (ks & 1) ? Bbuf0 : Bbuf1;
    if (ks + 1 < Ksteps) {
      stage(k0 + 32, nxt);            // loads for next slab overlap WMMA below
      __builtin_prefetch(Xrow + k0 + 64, 0, 1);  // global_prefetch_b8
    }
    const v16h a = cat8(*(const v8h*)(arow + k0), *(const v8h*)(arow + k0 + 16));
#pragma unroll
    for (int s = 0; s < 4; ++s) {
      const v16h b = loadA(&cur[(wn << 6) + (s << 4) + nloc][0], kbA);
      acc[s] = __builtin_amdgcn_wmma_f32_16x16x32_f16(false, a, false, b,
                                                      (short)0, acc[s], false, false);
    }
    __syncthreads();
  }

  // ---- epilogue: transpose through LDS, contiguous per-pixel stores ----
#pragma unroll
  for (int s = 0; s < 4; ++s) {
#pragma unroll
    for (int r = 0; r < 8; ++r) {
      float vv = acc[s][r] + bias[mW + r + hi];
      if (EPI == 1) vv = 0.5f * vv * (1.f + erff(vv * 0.70710678118654752f));
      const int colL = (wn << 6) + (s << 4) + nloc;  // pixel within tile
      const int rowL = (wm << 4) + r + hi;           // channel within tile
      if (EPI <= 1) sTh[colL][rowL] = (h16)vv;
      else          sTf[colL][rowL] = vv;
    }
  }
  __syncthreads();

  if (EPI <= 1) {
    h16* dst = Yh + (size_t)p * CSout + mBase;
#pragma unroll
    for (int i = 0; i < 4; ++i)
      *(v8h*)(dst + 8 * i) = *(const v8h*)&sTh[tid][8 * i];
  } else if (EPI == 2) {
    const float* rr = res + (size_t)p * CSout + mBase;
    float* dr = Yf + (size_t)p * CSout + mBase;
#pragma unroll
    for (int i = 0; i < 8; ++i) {
      v4f a0 = *(const v4f*)&sTf[tid][4 * i];
      v4f b0 = *(const v4f*)(rr + 4 * i);
      *(v4f*)(dr + 4 * i) = a0 + b0;
    }
  } else {  // EPI 3: final NCHW fp32 output
    const float* rr = res + (size_t)p * CSout + mBase;
#pragma unroll
    for (int j = 0; j < 32; ++j) {
      int row = mBase + j;
      if (row < M) Yf[(size_t)row * NPIX + p] = sTf[tid][j] + rr[j];
    }
  }
}

// ---------------------------------------------------------------------------
// Windowed self-attention. qkv: NHWC f16, stride 384, head-padded layout:
//   channel = branch*192 + {q:0|v:96} + head*16 + f, f<15 real, f=15 == 0.
// outs: NHWC f16, stride 192, channel = branch*96 + head*16 + f.
// Block = 128 thr (4 waves); each wave owns a 16-row strip of S.
// ---------------------------------------------------------------------------
template <int WS>
__global__ __launch_bounds__(128)
void attn_wmma(const h16* __restrict__ qkv, h16* __restrict__ outs,
               int branch, int sh, int sw) {
  constexpr int T  = WS * WS;
  constexpr int NT = T / 16;
  constexpr int NW = 256 / WS;

  __shared__ alignas(16) h16 qs[T][32];
  __shared__ alignas(16) h16 vsT[16][T];
  __shared__ alignas(16) h16 ps[4][16][T];

  const int tid  = threadIdx.x;
  const int wave = tid >> 5;
  const int lane = tid & 31;
  const int nloc = lane & 15;
  const int hi   = (lane & 16) ? 8 : 0;
  const int kbA  = (lane & 16) ? 8 : 0;
  const int kbB  = (lane & 16) ? 16 : 0;

  const int wy = blockIdx.x / NW, wx = blockIdx.x % NW;
  const int head = blockIdx.y;
  const int chq = branch * 192 + head * 16;        // 16B-aligned
  const int chv = chq + 96;
  const int ob  = branch * 96 + head * 16;

  // ---- stage q and v (two aligned b128 loads per token) ----
  for (int t = tid; t < T; t += 128) {
    int ty = t / WS, tx = t % WS;
    int oy = (wy * WS + ty - sh) & 255;
    int ox = (wx * WS + tx - sw) & 255;
    size_t po = (size_t)((oy << 8) + ox) * 384;
    v8h q0 = *(const v8h*)(qkv + po + chq);
    v8h q1 = *(const v8h*)(qkv + po + chq + 8);
    h16 qrow[32];
#pragma unroll
    for (int j = 0; j < 8; ++j) { qrow[j] = q0[j]; qrow[8 + j] = q1[j]; }
#pragma unroll
    for (int f = 16; f < 32; ++f) qrow[f] = (h16)0.f;
#pragma unroll
    for (int i = 0; i < 4; ++i)
      *(v8h*)&qs[t][8 * i] = *(const v8h*)&qrow[8 * i];
    v8h v0 = *(const v8h*)(qkv + po + chv);
    v8h v1 = *(const v8h*)(qkv + po + chv + 8);
#pragma unroll
    for (int j = 0; j < 8; ++j) { vsT[j][t] = v0[j]; vsT[8 + j][t] = v1[j]; }
  }
  __syncthreads();

  const int mrow = blockIdx.z * 64 + wave * 16;

  // ---- S = q q^T (single K step; f15/pad lanes are exact zeros) ----
  const v16h a = loadA(&qs[mrow + nloc][0], kbA);
  v8f sacc[NT];
#pragma unroll
  for (int t = 0; t < NT; ++t) sacc[t] = zero8();
#pragma unroll
  for (int t = 0; t < NT; ++t) {
    const v16h b = loadA(&qs[t * 16 + nloc][0], kbA);
    sacc[t] = __builtin_amdgcn_wmma_f32_16x16x32_f16(false, a, false, b,
                                                     (short)0, sacc[t], false, false);
  }

  // ---- in-register softmax; rows live across 16-lane halves ----
#pragma unroll
  for (int r = 0; r < 8; ++r) {
    float mx = -3.0e38f;
#pragma unroll
    for (int t = 0; t < NT; ++t) mx = fmaxf(mx, sacc[t][r]);
#pragma unroll
    for (int off = 1; off < 16; off <<= 1) mx = fmaxf(mx, __shfl_xor(mx, off, 32));
    float sum = 0.f;
#pragma unroll
    for (int t = 0; t < NT; ++t) {
      float e = __expf(sacc[t][r] - mx);
      sacc[t][r] = e;
      sum += e;
    }
#pragma unroll
    for (int off = 1; off < 16; off <<= 1) sum += __shfl_xor(sum, off, 32);
    float is = 1.f / sum;
#pragma unroll
    for (int t = 0; t < NT; ++t)
      ps[wave][r + hi][t * 16 + nloc] = (h16)(sacc[t][r] * is);
  }
  __syncthreads();

  // ---- O = P v ----
  v8f oacc = zero8();
  for (int kc = 0; kc < T; kc += 32) {
    const v16h pa = loadA(&ps[wave][nloc][kc], kbA);
    const v16h vb = loadA(&vsT[nloc][kc], kbA);
    oacc = __builtin_amdgcn_wmma_f32_16x16x32_f16(false, pa, false, vb,
                                                  (short)0, oacc, false, false);
  }

  // ---- store O into head-padded NHWC (f=15 column is pad, harmless) ----
#pragma unroll
  for (int r = 0; r < 8; ++r) {
    int tok = mrow + r + hi;
    int ty = tok / WS, tx = tok % WS;
    int oy = (wy * WS + ty - sh) & 255;
    int ox = (wx * WS + tx - sw) & 255;
    outs[(size_t)((oy << 8) + ox) * 192 + ob + nloc] = (h16)oacc[r];
  }
}

// ---------------------------------------------------------------------------
// x = x + BN(pixel_mixer(x) - x), NHWC fp32 stride 192 (thread = pixel)
// ---------------------------------------------------------------------------
__global__ __launch_bounds__(256)
void pixmix_bn_res(const float* __restrict__ x, const float* __restrict__ g,
                   const float* __restrict__ bb, const float* __restrict__ mm,
                   const float* __restrict__ vv, float* __restrict__ y) {
  int p = blockIdx.x * 256 + threadIdx.x;
  int h = p >> 8, w = p & 255;
#pragma unroll 5
  for (int ch = 0; ch < 180; ++ch) {
    int grp = ch % 5;
    int dy = 0, dx = 0;
    if (grp == 0) dx = 1;
    else if (grp == 1) dx = -1;
    else if (grp == 2) dy = 1;
    else if (grp == 3) dy = -1;
    int sp = (((h + dy) & 255) << 8) + ((w + dx) & 255);
    float mix = x[(size_t)sp * 192 + ch];
    float xv = x[(size_t)p * 192 + ch];
    float inv = g[ch] * rsqrtf(vv[ch] + EPS);
    y[(size_t)p * 192 + ch] = xv + (mix - xv) * inv + (bb[ch] - mm[ch] * inv);
  }
}

// ---------------------------------------------------------------------------
// Host orchestration
// ---------------------------------------------------------------------------
extern "C" void kernel_launch(void* const* d_in, const int* in_sizes, int n_in,
                              void* d_out, int out_size, void* d_ws, size_t ws_size,
                              hipStream_t stream) {
  const float* X0    = (const float*)d_in[0];
  const float* A0QW  = (const float*)d_in[1];
  const float* A0QB  = (const float*)d_in[2];
  const float* A0BNG = (const float*)d_in[3];
  const float* A0BNB = (const float*)d_in[4];
  const float* A0BNM = (const float*)d_in[5];
  const float* A0BNV = (const float*)d_in[6];
  const float* A0PW  = (const float*)d_in[7];
  const float* A0PB  = (const float*)d_in[8];
  const float* A1QW  = (const float*)d_in[9];
  const float* A1QB  = (const float*)d_in[10];
  const float* A1BNG = (const float*)d_in[11];
  const float* A1BNB = (const float*)d_in[12];
  const float* A1BNM = (const float*)d_in[13];
  const float* A1BNV = (const float*)d_in[14];
  const float* A1PW  = (const float*)d_in[15];
  const float* A1PB  = (const float*)d_in[16];
  const float* T2G   = (const float*)d_in[17];
  const float* T2B   = (const float*)d_in[18];
  const float* T2M   = (const float*)d_in[19];
  const float* T2V   = (const float*)d_in[20];
  const float* M0F1W = (const float*)d_in[21];
  const float* M0F1B = (const float*)d_in[22];
  const float* M0F2W = (const float*)d_in[23];
  const float* M0F2B = (const float*)d_in[24];
  const float* M1F1W = (const float*)d_in[25];
  const float* M1F1B = (const float*)d_in[26];
  const float* M1F2W = (const float*)d_in[27];
  const float* M1F2B = (const float*)d_in[28];
  const float* M2F1W = (const float*)d_in[29];
  const float* M2F1B = (const float*)d_in[30];
  const float* M2F2W = (const float*)d_in[31];
  const float* M2F2B = (const float*)d_in[32];

  char* wsb = (char*)d_ws;
  size_t off = 0;
  auto take = [&](size_t bytes) -> void* {
    void* pp = wsb + off;
    off += (bytes + 255) & ~(size_t)255;
    return pp;
  };

  h16* wqkv0  = (h16*)take((size_t)384 * 192 * sizeof(h16));
  h16* wqkv1  = (h16*)take((size_t)384 * 192 * sizeof(h16));
  h16* wproj0 = (h16*)take((size_t)192 * 192 * sizeof(h16));
  h16* wproj1 = (h16*)take((size_t)192 * 192 * sizeof(h16));
  h16* wfc1_0 = (h16*)take((size_t)384 * 192 * sizeof(h16));
  h16* wfc1_1 = (h16*)take((size_t)384 * 192 * sizeof(h16));
  h16* wfc1_2 = (h16*)take((size_t)384 * 192 * sizeof(h16));
  h16* wfc2_0 = (h16*)take((size_t)192 * 384 * sizeof(h16));
  h16* wfc2_1 = (h16*)take((size_t)192 * 384 * sizeof(h16));
  h16* wfc2_2 = (h16*)take((size_t)192 * 384 * sizeof(h16));
  float* fbq0 = (float*)take(384 * sizeof(float));
  float* fbq1 = (float*)take(384 * sizeof(float));
  float* pbp0 = (float*)take(192 * sizeof(float));
  float* pbp1 = (float*)take(192 * sizeof(float));
  float* pbf1_0 = (float*)take(384 * sizeof(float));
  float* pbf1_1 = (float*)take(384 * sizeof(float));
  float* pbf1_2 = (float*)take(384 * sizeof(float));
  float* pbf2_0 = (float*)take(192 * sizeof(float));
  float* pbf2_1 = (float*)take(192 * sizeof(float));
  float* pbf2_2 = (float*)take(192 * sizeof(float));
  float* xA   = (float*)take((size_t)NPIX * 192 * sizeof(float));
  float* xB   = (float*)take((size_t)NPIX * 192 * sizeof(float));
  h16* sh1    = (h16*)take((size_t)NPIX * 384 * sizeof(h16));  // qkv / hidden
  h16* outsb  = (h16*)take((size_t)NPIX * 192 * sizeof(h16));  // attention out

  // ---- prep: layouts, BN folds, padded biases ----
  to_nhwc<<<NPIX / 256, 256, 0, stream>>>(X0, xB);
  cvt_qkv<<<(384 * 192 + 255) / 256, 256, 0, stream>>>(A0QW, A0BNG, A0BNV, wqkv0);
  cvt_qkv<<<(384 * 192 + 255) / 256, 256, 0, stream>>>(A1QW, A1BNG, A1BNV, wqkv1);
  fuse_bias_qkv<<<2, 256, 0, stream>>>(A0QB, A0BNG, A0BNB, A0BNM, A0BNV, fbq0);
  fuse_bias_qkv<<<2, 256, 0, stream>>>(A1QB, A1BNG, A1BNB, A1BNM, A1BNV, fbq1);
  cvt_proj<<<(192 * 192 + 255) / 256, 256, 0, stream>>>(A0PW, wproj0);
  cvt_proj<<<(192 * 192 + 255) / 256, 256, 0, stream>>>(A1PW, wproj1);
  cvt_w_pad<<<(384 * 192 + 255) / 256, 256, 0, stream>>>(M0F1W, wfc1_0, 360, 180, 384, 192);
  cvt_w_pad<<<(384 * 192 + 255) / 256, 256, 0, stream>>>(M1F1W, wfc1_1, 360, 180, 384, 192);
  cvt_w_pad<<<(384 * 192 + 255) / 256, 256, 0, stream>>>(M2F1W, wfc1_2, 360, 180, 384, 192);
  cvt_w_pad<<<(192 * 384 + 255) / 256, 256, 0, stream>>>(M0F2W, wfc2_0, 180, 360, 192, 384);
  cvt_w_pad<<<(192 * 384 + 255) / 256, 256, 0, stream>>>(M1F2W, wfc2_1, 180, 360, 192, 384);
  cvt_w_pad<<<(192 * 384 + 255) / 256, 256, 0, stream>>>(M2F2W, wfc2_2, 180, 360, 192, 384);
  pad_bias<<<1, 256, 0, stream>>>(A0PB, pbp0, 180, 192);
  pad_bias<<<1, 256, 0, stream>>>(A1PB, pbp1, 180, 192);
  pad_bias<<<2, 256, 0, stream>>>(M0F1B, pbf1_0, 360, 384);
  pad_bias<<<2, 256, 0, stream>>>(M1F1B, pbf1_1, 360, 384);
  pad_bias<<<2, 256, 0, stream>>>(M2F1B, pbf1_2, 360, 384);
  pad_bias<<<1, 256, 0, stream>>>(M0F2B, pbf2_0, 180, 192);
  pad_bias<<<1, 256, 0, stream>>>(M1F2B, pbf2_1, 180, 192);
  pad_bias<<<1, 256, 0, stream>>>(M2F2B, pbf2_2, 180, 192);

  const dim3 gq(NPIX / 256, 12);  // 384 output channels
  const dim3 gd(NPIX / 256, 6);   // 192 output channels
  const dim3 ga8(1024, 6, 1);
  const dim3 ga16(256, 6, 4);

  // ---- block 0 (residual lives in xB) ----
  gemm_wmma<0, 0, float><<<gq, 256, 0, stream>>>(wqkv0, xB, fbq0, nullptr, nullptr, sh1,
                                                 384, 192, 192, 384, 1);
  attn_wmma<8><<<ga8, 128, 0, stream>>>(sh1, outsb, 0, 0, 0);
  attn_wmma<16><<<ga16, 128, 0, stream>>>(sh1, outsb, 1, 0, 0);
  gemm_wmma<2, 0, h16><<<gd, 256, 0, stream>>>(wproj0, outsb, pbp0, xB, xA, nullptr,
                                               192, 192, 192, 192, 1);
  gemm_wmma<1, 1, float><<<gq, 256, 0, stream>>>(wfc1_0, xA, pbf1_0, nullptr, nullptr, sh1,
                                                 384, 192, 192, 384, 36);
  gemm_wmma<2, 1, h16><<<gd, 256, 0, stream>>>(wfc2_0, sh1, pbf2_0, xA, xB, nullptr,
                                               192, 384, 384, 192, 72);

  // ---- block 1 (shifts 4,4 / 8,8) ----
  gemm_wmma<0, 0, float><<<gq, 256, 0, stream>>>(wqkv1, xB, fbq1, nullptr, nullptr, sh1,
                                                 384, 192, 192, 384, 1);
  attn_wmma<8><<<ga8, 128, 0, stream>>>(sh1, outsb, 0, 4, 4);
  attn_wmma<16><<<ga16, 128, 0, stream>>>(sh1, outsb, 1, 8, 8);
  gemm_wmma<2, 0, h16><<<gd, 256, 0, stream>>>(wproj1, outsb, pbp1, xB, xA, nullptr,
                                               192, 192, 192, 192, 1);
  gemm_wmma<1, 1, float><<<gq, 256, 0, stream>>>(wfc1_1, xA, pbf1_1, nullptr, nullptr, sh1,
                                                 384, 192, 192, 384, 36);
  gemm_wmma<2, 1, h16><<<gd, 256, 0, stream>>>(wfc2_1, sh1, pbf2_1, xA, xB, nullptr,
                                               192, 384, 384, 192, 72);

  // ---- pixel mixer + BN + residual ----
  pixmix_bn_res<<<NPIX / 256, 256, 0, stream>>>(xB, T2G, T2B, T2M, T2V, xA);

  // ---- MLP2, final residual straight into NCHW d_out ----
  gemm_wmma<1, 1, float><<<gq, 256, 0, stream>>>(wfc1_2, xA, pbf1_2, nullptr, nullptr, sh1,
                                                 384, 192, 192, 384, 36);
  gemm_wmma<3, 1, h16><<<gd, 256, 0, stream>>>(wfc2_2, sh1, pbf2_2, xA, (float*)d_out, nullptr,
                                               180, 384, 384, 192, 72);
  (void)in_sizes; (void)n_in; (void)out_size; (void)ws_size;
}
